// E_GCL_31825707663881
// MI455X (gfx1250) — matrix operand order repaired and checked by reading, not compiled, not run on hardware
//
#include <hip/hip_runtime.h>

typedef _Float16 v16h __attribute__((ext_vector_type(16)));
typedef _Float16 v8h  __attribute__((ext_vector_type(8)));
typedef float    v8f  __attribute__((ext_vector_type(8)));
typedef float    v4f  __attribute__((ext_vector_type(4)));

#define DCH 128     // feature dim
#define HST 136     // padded LDS row stride (f16) to break bank conflicts

// ---------------- prep kernels ----------------
__global__ void k_zero(float* __restrict__ p, int n) {
    int i = blockIdx.x * 256 + threadIdx.x;
    if (i < n) p[i] = 0.f;
}
__global__ void k_copy(const float* __restrict__ s, float* __restrict__ d, int n) {
    int i = blockIdx.x * 256 + threadIdx.x;
    if (i < n) d[i] = s[i];
}
__global__ void k_h2f16(const float* __restrict__ s, _Float16* __restrict__ d, int n) {
    int i = blockIdx.x * 256 + threadIdx.x;
    if (i < n) d[i] = (_Float16)s[i];
}
// Pack row-major [K x 128] f32 weight into per-lane WMMA B-fragment order:
// idx = ((ntile*ksteps + kstep)*32 + lane)*16 + j ; n = ntile*16 + lane%16 ;
// k = kstep*32 + (lane/16)*8 + (j&7) + (j/8)*16   (matches V_WMMA 32x16 f16 B layout)
__global__ void k_pack(const float* __restrict__ W, _Float16* __restrict__ Wp, int ksteps) {
    int i = blockIdx.x * 256 + threadIdx.x;
    int total = 8 * ksteps * 512;
    if (i >= total) return;
    int j = i & 15;
    int l = (i >> 4) & 31;
    int q = i >> 9;
    int s = q % ksteps;
    int t = q / ksteps;
    int n = t * 16 + (l & 15);
    int k = s * 32 + (l >> 4) * 8 + (j & 7) + (j >> 3) * 16;
    Wp[i] = (_Float16)W[k * DCH + n];
}

// A fragment (16x32 f16): lane m = lane%16; half = lane/16 selects the two
// 8-element K runs at koff and koff+16 within the row (ISA 7.12.2 layout).
__device__ __forceinline__ v16h load_a(const _Float16* row, int koff) {
    v8h lo = *(const v8h*)(row + koff);
    v8h hi = *(const v8h*)(row + koff + 16);
    return __builtin_shufflevector(lo, hi, 0,1,2,3,4,5,6,7,8,9,10,11,12,13,14,15);
}

// ---------------- fused edge MLP + scatter-add ----------------
__launch_bounds__(256, 2)
__global__ void k_edge(const _Float16* __restrict__ Hh,
                       const float* __restrict__ pos,
                       const int* __restrict__ ei,
                       const _Float16* __restrict__ W1p,
                       const float* __restrict__ We1,   // raw, for radial row 256
                       const float* __restrict__ b1,
                       const _Float16* __restrict__ W2p,
                       const float* __restrict__ b2,
                       float* __restrict__ agg,
                       int Me)
{
    __shared__ _Float16 Hbuf[128 * HST];  // rows 0..63: h[row], 64..127: h[col]
    __shared__ _Float16 Hid[64 * HST];    // layer-1 output
    __shared__ int   ridx[64];
    __shared__ int   cidx[64];
    __shared__ float rad[64];

    const int tid = threadIdx.x;
    const int e0  = blockIdx.x * 64;

    if (tid < 64) {
        int e = e0 + tid;
        int r = 0, c = 0; float rr = 0.f;
        if (e < Me) {
            r = ei[e]; c = ei[Me + e];
            float dx = pos[3*r+0] - pos[3*c+0];
            float dy = pos[3*r+1] - pos[3*c+1];
            float dz = pos[3*r+2] - pos[3*c+2];
            rr = dx*dx + dy*dy + dz*dz;
        }
        ridx[tid] = r; cidx[tid] = c; rad[tid] = rr;
    }
    __syncthreads();

    // gather 128 rows x 128 f16 (16B chunks), 8 chunks per thread
    #pragma unroll
    for (int i = 0; i < 8; ++i) {
        int c2  = tid + 256 * i;
        int row = c2 >> 4;
        int j   = c2 & 15;
        int src = (row < 64) ? ridx[row] : cidx[row - 64];
        *(v8h*)(&Hbuf[row * HST + j * 8]) = *(const v8h*)(Hh + (size_t)src * DCH + j * 8);
    }
    __syncthreads();

    const int wave = tid >> 5, lane = tid & 31;
    const int nl = lane & 15, half = lane >> 4;
    const int n  = wave * 16 + nl;

    const float wr_n = We1[256 * DCH + n];  // radial weight row
    const float b1_n = b1[n];
    const float b2_n = b2[n];

    v8f acc[4];
    #pragma unroll
    for (int mt = 0; mt < 4; ++mt)
        #pragma unroll
        for (int q = 0; q < 8; ++q) acc[mt][q] = 0.f;

    // layer 1: K = 256 (128 from h[row], 128 from h[col])
    #pragma unroll
    for (int s = 0; s < 8; ++s) {
        v16h b = *(const v16h*)(W1p + (((size_t)wave * 8 + s) * 32 + lane) * 16);
        const int kb     = s * 32;
        const int rowoff = (kb < 128) ? 0 : 64;
        const int koff   = (kb & 127) + half * 8;
        #pragma unroll
        for (int mt = 0; mt < 4; ++mt) {
            v16h a = load_a(&Hbuf[(rowoff + mt * 16 + nl) * HST], koff);
            acc[mt] = __builtin_amdgcn_wmma_f32_16x16x32_f16(
                false, a, false, b, (short)0, acc[mt], false, false);
        }
    }

    // epilogue 1: + radial*wr + bias, ReLU, stash to LDS as f16
    #pragma unroll
    for (int mt = 0; mt < 4; ++mt) {
        #pragma unroll
        for (int r = 0; r < 8; ++r) {
            int m = mt * 16 + r + 8 * half;
            float v = acc[mt][r] + rad[m] * wr_n + b1_n;
            v = v > 0.f ? v : 0.f;
            Hid[m * HST + n] = (_Float16)v;
            acc[mt][r] = 0.f;
        }
    }
    __syncthreads();

    // layer 2: K = 128
    #pragma unroll
    for (int s = 0; s < 4; ++s) {
        v16h b = *(const v16h*)(W2p + (((size_t)wave * 4 + s) * 32 + lane) * 16);
        const int koff = s * 32 + half * 8;
        #pragma unroll
        for (int mt = 0; mt < 4; ++mt) {
            v16h a = load_a(&Hid[(mt * 16 + nl) * HST], koff);
            acc[mt] = __builtin_amdgcn_wmma_f32_16x16x32_f16(
                false, a, false, b, (short)0, acc[mt], false, false);
        }
    }

    // epilogue 2: bias, ReLU, scatter-add into agg (f32 HW atomics, L2-resident)
    #pragma unroll
    for (int mt = 0; mt < 4; ++mt) {
        #pragma unroll
        for (int r = 0; r < 8; ++r) {
            int m = mt * 16 + r + 8 * half;
            if (e0 + m < Me) {
                float v = acc[mt][r] + b2_n;
                v = v > 0.f ? v : 0.f;
                unsafeAtomicAdd(&agg[(size_t)ridx[m] * DCH + n], v);
            }
        }
    }
}

// ---------------- fused node MLP + residual ----------------
__launch_bounds__(256, 2)
__global__ void k_node(const _Float16* __restrict__ Hh,
                       const float* __restrict__ hsrc,
                       const _Float16* __restrict__ W1p,
                       const float* __restrict__ b1,
                       const _Float16* __restrict__ W2p,
                       const float* __restrict__ b2,
                       float* __restrict__ out,   // in: agg (f32), out: h_out
                       int Nn)
{
    __shared__ _Float16 Hbuf[128 * HST];  // rows 0..63: h (f16), 64..127: agg (f16)
    __shared__ _Float16 Hid[64 * HST];

    const int tid = threadIdx.x;
    const int n0  = blockIdx.x * 64;

    #pragma unroll
    for (int i = 0; i < 4; ++i) {          // stage h rows
        int c2  = tid + 256 * i;
        int row = c2 >> 4, j = c2 & 15;
        int src = n0 + row; if (src >= Nn) src = Nn - 1;
        *(v8h*)(&Hbuf[row * HST + j * 8]) = *(const v8h*)(Hh + (size_t)src * DCH + j * 8);
    }
    #pragma unroll
    for (int i = 0; i < 8; ++i) {          // stage agg rows (f32 -> f16)
        int c2  = tid + 256 * i;
        int row = c2 >> 5, j = c2 & 31;
        int src = n0 + row; if (src >= Nn) src = Nn - 1;
        v4f v = *(const v4f*)(out + (size_t)src * DCH + j * 4);
        _Float16* dp = &Hbuf[(64 + row) * HST + j * 4];
        dp[0] = (_Float16)v[0]; dp[1] = (_Float16)v[1];
        dp[2] = (_Float16)v[2]; dp[3] = (_Float16)v[3];
    }
    __syncthreads();

    const int wave = tid >> 5, lane = tid & 31;
    const int nl = lane & 15, half = lane >> 4;
    const int n  = wave * 16 + nl;
    const float b1_n = b1[n];
    const float b2_n = b2[n];

    v8f acc[4];
    #pragma unroll
    for (int mt = 0; mt < 4; ++mt)
        #pragma unroll
        for (int q = 0; q < 8; ++q) acc[mt][q] = 0.f;

    #pragma unroll
    for (int s = 0; s < 8; ++s) {          // layer 1: K = 256 ([h, agg])
        v16h b = *(const v16h*)(W1p + (((size_t)wave * 8 + s) * 32 + lane) * 16);
        const int kb     = s * 32;
        const int rowoff = (kb < 128) ? 0 : 64;
        const int koff   = (kb & 127) + half * 8;
        #pragma unroll
        for (int mt = 0; mt < 4; ++mt) {
            v16h a = load_a(&Hbuf[(rowoff + mt * 16 + nl) * HST], koff);
            acc[mt] = __builtin_amdgcn_wmma_f32_16x16x32_f16(
                false, a, false, b, (short)0, acc[mt], false, false);
        }
    }

    #pragma unroll
    for (int mt = 0; mt < 4; ++mt) {
        #pragma unroll
        for (int r = 0; r < 8; ++r) {
            int m = mt * 16 + r + 8 * half;
            float v = acc[mt][r] + b1_n;
            v = v > 0.f ? v : 0.f;
            Hid[m * HST + n] = (_Float16)v;
            acc[mt][r] = 0.f;
        }
    }
    __syncthreads();

    #pragma unroll
    for (int s = 0; s < 4; ++s) {          // layer 2: K = 128, no ReLU
        v16h b = *(const v16h*)(W2p + (((size_t)wave * 4 + s) * 32 + lane) * 16);
        const int koff = s * 32 + half * 8;
        #pragma unroll
        for (int mt = 0; mt < 4; ++mt) {
            v16h a = load_a(&Hid[(mt * 16 + nl) * HST], koff);
            acc[mt] = __builtin_amdgcn_wmma_f32_16x16x32_f16(
                false, a, false, b, (short)0, acc[mt], false, false);
        }
    }

    #pragma unroll
    for (int mt = 0; mt < 4; ++mt) {
        #pragma unroll
        for (int r = 0; r < 8; ++r) {
            int m  = mt * 16 + r + 8 * half;
            int gm = n0 + m;
            if (gm < Nn)
                out[(size_t)gm * DCH + n] =
                    hsrc[(size_t)gm * DCH + n] + acc[mt][r] + b2_n;
        }
    }
}

// ---------------- launcher ----------------
extern "C" void kernel_launch(void* const* d_in, const int* in_sizes, int n_in,
                              void* d_out, int out_size, void* d_ws, size_t ws_size,
                              hipStream_t stream)
{
    const float* h    = (const float*)d_in[0];
    const float* pos  = (const float*)d_in[1];
    const float* We1  = (const float*)d_in[2];
    const float* be1  = (const float*)d_in[3];
    const float* We2  = (const float*)d_in[4];
    const float* be2  = (const float*)d_in[5];
    const float* Wn1  = (const float*)d_in[6];
    const float* bn1  = (const float*)d_in[7];
    const float* Wn2  = (const float*)d_in[8];
    const float* bn2  = (const float*)d_in[9];
    const int*   ei   = (const int*)d_in[10];

    const int Nn = in_sizes[0] / DCH;
    const int Me = in_sizes[10] / 2;
    float* out = (float*)d_out;

    // workspace layout: Hh (f16 h), then 4 packed weight blobs (~12.4 MiB total)
    char* ws = (char*)d_ws;
    size_t off = 0;
    _Float16* Hh   = (_Float16*)(ws + off); off += (size_t)Nn * DCH * 2; off = (off + 255) & ~(size_t)255;
    _Float16* W1p  = (_Float16*)(ws + off); off += 8 * 8 * 512 * 2;   // K=256
    _Float16* W2p  = (_Float16*)(ws + off); off += 8 * 4 * 512 * 2;   // K=128
    _Float16* Wn1p = (_Float16*)(ws + off); off += 8 * 8 * 512 * 2;
    _Float16* Wn2p = (_Float16*)(ws + off); off += 8 * 4 * 512 * 2;

    // prep: zero agg region of out, copy positions tail, h->f16, pack weights
    k_zero <<<(Nn * DCH + 255) / 256, 256, 0, stream>>>(out, Nn * DCH);
    k_copy <<<(Nn * 3   + 255) / 256, 256, 0, stream>>>(pos, out + (size_t)Nn * DCH, Nn * 3);
    k_h2f16<<<(Nn * DCH + 255) / 256, 256, 0, stream>>>(h, Hh, Nn * DCH);
    k_pack <<<(8 * 8 * 512 + 255) / 256, 256, 0, stream>>>(We1, W1p, 8);
    k_pack <<<(8 * 4 * 512 + 255) / 256, 256, 0, stream>>>(We2, W2p, 4);
    k_pack <<<(8 * 8 * 512 + 255) / 256, 256, 0, stream>>>(Wn1, Wn1p, 8);
    k_pack <<<(8 * 4 * 512 + 255) / 256, 256, 0, stream>>>(Wn2, Wn2p, 4);

    // fused edge MLP + scatter-add into out[0 .. Nn*128)
    k_edge<<<(Me + 63) / 64, 256, 0, stream>>>(Hh, pos, ei, W1p, We1, be1, W2p, be2, out, Me);
    // fused node MLP + residual (reads agg from out, overwrites with h_out)
    k_node<<<(Nn + 63) / 64, 256, 0, stream>>>(Hh, h, Wn1p, bn1, Wn2p, bn2, out, Nn);
}
